// WeightAdaptiveGallinear_37915971289186
// MI455X (gfx1250) — matrix operand order repaired and checked by reading, not compiled, not run on hardware
//
#include <hip/hip_runtime.h>
#include <hip/hip_bf16.h>

typedef __attribute__((ext_vector_type(8)))  float  v8f;
typedef __attribute__((ext_vector_type(16))) __bf16 v16bf;
typedef __attribute__((ext_vector_type(8)))  __bf16 v8bf;

#define TILE_M 64
#define KC     64
#define NJ     128
#define XROW   18   // IN_FEATURES + 1 + LATENT

// ---------------------------------------------------------------------------
// Kernel A: fold the harmonic basis into fc2 once per call.
//   W_eff[j,c] = sum_k basis[k] * fc2_w[(j*32+k)*4096 + c]   (stored bf16,
//   pre-blocked per K-chunk of 64 as [chunk][j:128][kl:64] for contiguous
//   async staging), b_eff[j] = sum_k basis[k]*fc2_b[j*32+k] (f32).
// ---------------------------------------------------------------------------
__global__ __launch_bounds__(256)
void build_weff(const float* __restrict__ x, const float* __restrict__ fc2_w,
                const float* __restrict__ fc2_b, const float* __restrict__ dil,
                const float* __restrict__ shf,
                __hip_bfloat16* __restrict__ wq_, float* __restrict__ beff)
{
    __bf16* wq = reinterpret_cast<__bf16*>(wq_);
    const int tid = threadIdx.x;
    const int j   = blockIdx.y;               // 0..127
    const int c   = blockIdx.x * 256 + tid;   // 0..4095

    const float s   = x[(size_t)16383 * XROW + 1];
    const float arg = s * dil[0] + shf[0];
    float basis[32];
#pragma unroll
    for (int k = 0; k < 16; ++k) {
        float a = arg * ((float)k * (16.0f / 15.0f));  // linspace(0,16,16)
        basis[k]      = __cosf(a);
        basis[16 + k] = __sinf(a);
    }

    float acc = 0.f;
#pragma unroll
    for (int k = 0; k < 32; ++k)
        acc += basis[k] * fc2_w[(size_t)(j * 32 + k) * 4096 + c];

    wq[((size_t)(c >> 6) * NJ + j) * KC + (c & 63)] = (__bf16)acc;

    if (blockIdx.x == 0 && tid == 0) {
        float b = 0.f;
#pragma unroll
        for (int k = 0; k < 32; ++k) b += basis[k] * fc2_b[j * 32 + k];
        beff[j] = b;
    }
}

// ---------------------------------------------------------------------------
// Kernel B: fused  h = relu(z@fc1_wT + fc1_b)  ->  w = h @ W_effT + b_eff
//           -> out = inp*(w[:, :64]) + w[:, 64:128]
// 256 threads (8 wave32), 64 batch rows per block. K-loop over 4096 in
// chunks of 64: A (h) computed on VALU into LDS, B (W_eff) async-staged
// into LDS, MACs on v_wmma_f32_16x16x32_bf16, f32 accumulate.
// ---------------------------------------------------------------------------
__global__ __launch_bounds__(256)
void fused_gal(const float* __restrict__ x, const float* __restrict__ fc1_w,
               const float* __restrict__ fc1_b,
               const __hip_bfloat16* __restrict__ wq_,
               const float* __restrict__ beff, float* __restrict__ out)
{
    const __bf16* wq = reinterpret_cast<const __bf16*>(wq_);

    __shared__ __bf16 hlds[TILE_M][72];   // A chunk: 64 rows x 64 k (+pad)
    __shared__ __bf16 wlds[NJ][72];       // B chunk: 128 j  x 64 k (+pad)
    __shared__ float  inp_s[TILE_M];
    __shared__ float  beff_s[128];

    const int tid  = threadIdx.x;
    const int lane = tid & 31;
    const int wv   = tid >> 5;
    const int m0   = blockIdx.x * TILE_M;

    if (tid < TILE_M) inp_s[tid] = x[(size_t)(m0 + tid) * XROW];
    if (tid < 128)    beff_s[tid] = beff[tid];

    // each thread owns one batch row (r) and a quarter of the k-chunk (kq)
    const int r  = tid >> 2;
    const int kq = tid & 3;
    float z[16];
#pragma unroll
    for (int jj = 0; jj < 16; ++jj)
        z[jj] = x[(size_t)(m0 + r) * XROW + 2 + jj];

    const int mloc = (wv & 3) * 16;   // wave's 16-row M sub-tile
    const int nh   = wv >> 2;         // wave's N half
    const int hl   = lane >> 4;
    const int ln   = lane & 15;

    v8f acc[4];
#pragma unroll
    for (int t = 0; t < 4; ++t)
#pragma unroll
        for (int e = 0; e < 8; ++e) acc[t][e] = 0.f;

    __syncthreads();

    for (int kc = 0; kc < 4096; kc += KC) {
        // ---- stage B chunk (16KB contiguous) async global -> LDS ----------
        {
            const __bf16* gsrc = wq + (size_t)(kc >> 6) * (NJ * KC);
#pragma unroll
            for (int i = 0; i < 4; ++i) {
                int e   = tid + 256 * i;        // 0..1023 x 16B
                int row = e >> 3;
                int seg = e & 7;
                unsigned ldst =
                    (unsigned)(size_t)(const void*)&wlds[row][seg * 8];
                unsigned long long ga =
                    (unsigned long long)(size_t)(const void*)(gsrc + e * 8);
                asm volatile("global_load_async_to_lds_b128 %0, %1, off"
                             :: "v"(ldst), "v"(ga) : "memory");
            }
        }

        // ---- compute A chunk: h = relu(z . fc1_w + fc1_b), K=16 on VALU ---
#pragma unroll 4
        for (int i = 0; i < 16; ++i) {
            int kg = kc + kq * 16 + i;
            const float4* fw = (const float4*)(fc1_w + (size_t)kg * 16);
            float4 f0 = fw[0], f1 = fw[1], f2 = fw[2], f3 = fw[3];
            float a = fc1_b[kg];
            a += z[0]  * f0.x + z[1]  * f0.y + z[2]  * f0.z + z[3]  * f0.w;
            a += z[4]  * f1.x + z[5]  * f1.y + z[6]  * f1.z + z[7]  * f1.w;
            a += z[8]  * f2.x + z[9]  * f2.y + z[10] * f2.z + z[11] * f2.w;
            a += z[12] * f3.x + z[13] * f3.y + z[14] * f3.z + z[15] * f3.w;
            hlds[r][kq * 16 + i] = (__bf16)fmaxf(a, 0.f);
        }
        if (kc + KC < 4096)
            __builtin_prefetch(fc1_w + (size_t)(kc + KC) * 16, 0, 0);

#if __has_builtin(__builtin_amdgcn_s_wait_asynccnt)
        __builtin_amdgcn_s_wait_asynccnt(0);
#else
        asm volatile("s_wait_asynccnt 0" ::: "memory");
#endif
        __syncthreads();

        // ---- WMMA: 2 K-steps of 32 x 4 N-tiles ----------------------------
#pragma unroll
        for (int kb = 0; kb < KC; kb += 32) {
            union { v16bf v; v8bf h[2]; } A;
            A.h[0] = *(const v8bf*)&hlds[mloc + ln][kb + 8 * hl];
            A.h[1] = *(const v8bf*)&hlds[mloc + ln][kb + 16 + 8 * hl];
#pragma unroll
            for (int ti = 0; ti < 4; ++ti) {
                int t = 2 * nh + (ti & 1) + 4 * (ti >> 1); // {2nh,2nh+1,2nh+4,2nh+5}
                int jc = t * 16 + ln;
                union { v16bf v; v8bf h[2]; } Bm;
                Bm.h[0] = *(const v8bf*)&wlds[jc][kb + 8 * hl];
                Bm.h[1] = *(const v8bf*)&wlds[jc][kb + 16 + 8 * hl];
                acc[ti] = __builtin_amdgcn_wmma_f32_16x16x32_bf16(
                    false, A.v, false, Bm.v, (short)0, acc[ti], false, false);
            }
        }
        __syncthreads();
    }

    // ---- epilogue: out[b,o] = inp[b]*(w[b,o]+be[o]) + (w[b,o+64]+be[o+64])
#pragma unroll
    for (int ti = 0; ti < 2; ++ti) {
        int t = 2 * nh + ti;          // weight tile 0..3
        int o = t * 16 + ln;          // 0..63
        float bw = beff_s[o];
        float bb = beff_s[o + 64];
        v8f wacc = acc[ti];
        v8f bacc = acc[ti + 2];       // matching bias tile (t+4)
#pragma unroll
        for (int v = 0; v < 8; ++v) {
            int mrow = mloc + v + 8 * hl;
            float iv = inp_s[mrow];
            out[(size_t)(m0 + mrow) * 64 + o] =
                iv * (wacc[v] + bw) + (bacc[v] + bb);
        }
    }
}

// ---------------------------------------------------------------------------
extern "C" void kernel_launch(void* const* d_in, const int* in_sizes, int n_in,
                              void* d_out, int out_size, void* d_ws, size_t ws_size,
                              hipStream_t stream)
{
    (void)in_sizes; (void)n_in; (void)out_size; (void)ws_size;
    const float* x      = (const float*)d_in[0];
    const float* fc1_w  = (const float*)d_in[1];
    const float* fc1_b  = (const float*)d_in[2];
    const float* fc2_w  = (const float*)d_in[3];
    const float* fc2_b  = (const float*)d_in[4];
    const float* dil    = (const float*)d_in[5];
    const float* shf    = (const float*)d_in[6];

    __hip_bfloat16* wq = (__hip_bfloat16*)d_ws;                  // 1 MB bf16
    float* beff        = (float*)((char*)d_ws + (1u << 20));     // 512 B f32
    float* out         = (float*)d_out;

    dim3 gA(16, 128);
    build_weff<<<gA, 256, 0, stream>>>(x, fc2_w, fc2_b, dil, shf, wq, beff);
    fused_gal<<<256, 256, 0, stream>>>(x, fc1_w, fc1_b, wq, beff, out);
}
